// Encoder_25924422599298
// MI455X (gfx1250) — matrix operand order repaired and checked by reading, not compiled, not run on hardware
//
#include <hip/hip_runtime.h>
#include <math.h>

typedef __bf16 bf16;
typedef __attribute__((ext_vector_type(16))) __bf16 v16bf;
typedef __attribute__((ext_vector_type(8)))  __bf16 v8bf;
typedef __attribute__((ext_vector_type(8)))  float  v8f;
typedef __attribute__((ext_vector_type(4)))  int    v4i;

#define DIM 768
#define DIMQ 64
#define NHEAD 12
#define SEQ 2048
#define BATCH 4
#define MTOT (BATCH * SEQ)   // 8192
#define QKVS 256             // padded q|k|v row stride (192 -> 256)

// -------------------- CDNA5 async-to-LDS staging (guarded) --------------------

#if defined(__has_builtin)
#if __has_builtin(__builtin_amdgcn_global_load_async_to_lds_b128)
#define HAVE_ASYNC 1
#endif
#endif

#ifdef HAVE_ASYNC
#if __has_builtin(__builtin_amdgcn_s_wait_asynccnt)
#define WAIT_ASYNC(n) __builtin_amdgcn_s_wait_asynccnt(n)
#else
#define WAIT_ASYNC(n) asm volatile("s_wait_asynccnt %0" :: "i"(n) : "memory")
#endif
#else
#define WAIT_ASYNC(n) ((void)0)
#endif

__device__ inline void copy16(const bf16* g, bf16* l) {
#ifdef HAVE_ASYNC
  __builtin_amdgcn_global_load_async_to_lds_b128(
      (__attribute__((address_space(1))) v4i*)g,
      (__attribute__((address_space(3))) v4i*)l, 0, 0);
#else
  *(int4*)l = *(const int4*)g;
#endif
}

// -------------------- WMMA helpers --------------------

__device__ inline v16bf cat8(v8bf lo, v8bf hi) {
  return __builtin_shufflevector(lo, hi, 0,1,2,3,4,5,6,7,8,9,10,11,12,13,14,15);
}

__device__ inline v8f wmma_bf16(v16bf a, v16bf b, v8f c) {
  return __builtin_amdgcn_wmma_f32_16x16x32_bf16(false, a, false, b, (short)0, c, false, false);
}

// A-fragment (16x32 bf16) from LDS, row-major, element stride `ldl`.
__device__ inline v16bf load_a_frag(const bf16* base, int ldl, int lane) {
  int row = lane & 15;
  int kc  = (lane >> 4) << 3;
  v8bf lo = *(const v8bf*)(base + row * ldl + kc);
  v8bf hi = *(const v8bf*)(base + row * ldl + 16 + kc);
  return cat8(lo, hi);
}

// B-fragment (32x16 bf16) from LDS stored TRANSPOSED [n][k], stride `ldl`.
__device__ inline v16bf load_b_frag(const bf16* base, int ldl, int lane) {
  int col = lane & 15;
  int kk  = (lane >> 4) << 3;
  v8bf lo = *(const v8bf*)(base + col * ldl + kk);
  v8bf hi = *(const v8bf*)(base + col * ldl + 16 + kk);
  return cat8(lo, hi);
}

// -------------------- weight prep (all weights stored TRANSPOSED [N][K]) ----

__global__ void pack_qkv_w(const float* __restrict__ wq, const float* __restrict__ wk,
                           const float* __restrict__ wv, bf16* __restrict__ out) {
  int idx = blockIdx.x * 256 + threadIdx.x;          // QKVS*DIM
  if (idx >= QKVS * DIM) return;
  int c = idx / DIM, d = idx % DIM;
  float v = (c < 64)  ? wq[d * 64 + c]
          : (c < 128) ? wk[d * 64 + (c - 64)]
          : (c < 192) ? wv[d * 64 + (c - 128)] : 0.f;
  out[idx] = (bf16)v;
}

// fold lin_w [768,768] -> transposed [768][64]: W'[n][d] = sum_j lin_w[j*64+d, n]
__global__ void fold_lin_w(const float* __restrict__ lw, bf16* __restrict__ out) {
  int idx = blockIdx.x * 256 + threadIdx.x;          // DIM*64
  if (idx >= DIM * 64) return;
  int n = idx / 64, d = idx % 64;
  float s = 0.f;
  #pragma unroll
  for (int j = 0; j < NHEAD; ++j) s += lw[(size_t)(j * 64 + d) * DIM + n];
  out[idx] = (bf16)s;
}

// transpose+convert: in [K][N] f32 -> out [N][K] bf16
__global__ void cvt_t_bf16(const float* __restrict__ in, bf16* __restrict__ out,
                           int K, int N) {
  int idx = blockIdx.x * 256 + threadIdx.x;
  if (idx >= N * K) return;
  int n = idx / K, k = idx % K;
  out[idx] = (bf16)in[(size_t)k * N + n];
}

// -------------------- layernorm: one row (768) per block --------------------

__global__ __launch_bounds__(256)
void layernorm_kernel(const float* __restrict__ x, const float* __restrict__ g,
                      const float* __restrict__ b, bf16* __restrict__ out) {
  __shared__ float wsum[8], wsq[8];
  int row = blockIdx.x, tid = threadIdx.x;
  const float* xr = x + (size_t)row * DIM;
  float v[3], s = 0.f, sq = 0.f;
  #pragma unroll
  for (int j = 0; j < 3; ++j) { v[j] = xr[j * 256 + tid]; s += v[j]; sq += v[j] * v[j]; }
  #pragma unroll
  for (int m = 1; m < 32; m <<= 1) { s += __shfl_xor(s, m); sq += __shfl_xor(sq, m); }
  int wave = tid >> 5, lane = tid & 31;
  if (lane == 0) { wsum[wave] = s; wsq[wave] = sq; }
  __syncthreads();
  s = 0.f; sq = 0.f;
  #pragma unroll
  for (int w = 0; w < 8; ++w) { s += wsum[w]; sq += wsq[w]; }
  float mu = s * (1.f / DIM);
  float var = sq * (1.f / DIM) - mu * mu;
  float rs = rsqrtf(var + 1e-5f);
  #pragma unroll
  for (int j = 0; j < 3; ++j) {
    int c = j * 256 + tid;
    out[(size_t)row * DIM + c] = (bf16)((v[j] - mu) * rs * g[c] + b[c]);
  }
}

// -------------------- tiled bf16 GEMM, double-buffered async WMMA core -----
// C[M,N] = A[M,K] * Bt[N,K]^T. Block 128x128, BLOCK_K=64, 256 thr = 8 waves.
// Wave grid 4x2: each wave computes 32x64 (2x4 WMMA tiles, 16 wmma/stage).
// EPI 1: +bias +resid -> f32. EPI 2: +bias, exact GELU -> bf16.
// EPI 3: store bf16 (stride NS) + scatter V columns transposed into aux.

#define LDA 72   // padded LDS stride (bf16): 144B = 36 dwords -> conflict-free

template<int EPI>
__global__ __launch_bounds__(256)
void gemm_bf16(const bf16* __restrict__ A, const bf16* __restrict__ Bt,
               const float* __restrict__ bias, const float* __restrict__ resid,
               void* __restrict__ Cout, void* __restrict__ aux,
               int M, int N, int K, int NS) {
  __shared__ bf16 As[2][128 * LDA];
  __shared__ bf16 Bs[2][128 * LDA];    // transposed weights: [n][k]
  int tid = threadIdx.x;
  int wave = tid >> 5, lane = tid & 31;
  int wm = wave >> 1, wn = wave & 1;   // wm 0..3 (32 rows), wn 0..1 (64 cols)
  int m0 = blockIdx.y * 128, n0 = blockIdx.x * 128;
  int row = tid >> 1, hf2 = (tid & 1) * 32;

  const bf16* baseA = A  + (size_t)(m0 + row) * K + hf2;
  const bf16* baseB = Bt + (size_t)(n0 + row) * K + hf2;

  auto issue = [&](int k0, int buf) {
    bf16* dstA = &As[buf][row * LDA + hf2];
    bf16* dstB = &Bs[buf][row * LDA + hf2];
    #pragma unroll
    for (int j = 0; j < 4; ++j) {
      copy16(baseA + k0 + j * 8, dstA + j * 8);
      copy16(baseB + k0 + j * 8, dstB + j * 8);
    }
  };

  v8f acc[2][4] = {};
  int nstages = K >> 6;
  issue(0, 0);

  for (int s = 0; s < nstages; ++s) {
    int buf = s & 1;
    if (s + 1 < nstages) {          // software pipeline: prefetch next stage
      issue((s + 1) << 6, buf ^ 1);
      WAIT_ASYNC(8);                // 8 younger ops pending; current stage done
    } else {
      WAIT_ASYNC(0);
    }
    __syncthreads();

    #pragma unroll
    for (int kk = 0; kk < 2; ++kk) {
      v16bf af[2], bfr[4];
      #pragma unroll
      for (int ti = 0; ti < 2; ++ti)
        af[ti] = load_a_frag(&As[buf][(wm * 32 + ti * 16) * LDA + kk * 32], LDA, lane);
      #pragma unroll
      for (int tj = 0; tj < 4; ++tj)
        bfr[tj] = load_b_frag(&Bs[buf][(wn * 64 + tj * 16) * LDA + kk * 32], LDA, lane);
      #pragma unroll
      for (int ti = 0; ti < 2; ++ti)
        #pragma unroll
        for (int tj = 0; tj < 4; ++tj)
          acc[ti][tj] = wmma_bf16(af[ti], bfr[tj], acc[ti][tj]);
    }
    __syncthreads();                // all waves done reading buf before reuse
  }

  int hf = lane >> 4, ln = lane & 15;
  #pragma unroll
  for (int ti = 0; ti < 2; ++ti)
    #pragma unroll
    for (int tj = 0; tj < 4; ++tj)
      #pragma unroll
      for (int i = 0; i < 8; ++i) {
        int m = m0 + wm * 32 + ti * 16 + i + hf * 8;
        int n = n0 + wn * 64 + tj * 16 + ln;
        float v = acc[ti][tj][i];
        if (EPI == 1) {
          size_t o = (size_t)m * N + n;
          ((float*)Cout)[o] = v + bias[n] + resid[o];
        } else if (EPI == 2) {
          v += bias[n];
          v = 0.5f * v * (1.0f + erff(v * 0.70710678118654752f));
          ((bf16*)Cout)[(size_t)m * N + n] = (bf16)v;
        } else { // EPI 3: qkv store + transposed V sideband
          ((bf16*)Cout)[(size_t)m * NS + n] = (bf16)v;
          if (n >= 128 && n < 192) {
            int b = m >> 11, sdx = m & (SEQ - 1);
            ((bf16*)aux)[((size_t)b * DIMQ + (n - 128)) * SEQ + sdx] = (bf16)v;
          }
        }
      }
}

// -------------------- flash attention --------------------
// qkv: [B][S][QKVS] bf16 (q|k|..). vT: [B][64][S] bf16 (V transposed).
// One block = 64 query rows, 4 waves x 16 rows. All LDS staging contiguous/async.

#define LDT 72

__global__ __launch_bounds__(128)
void attn_kernel(const bf16* __restrict__ qkv, const bf16* __restrict__ vT,
                 bf16* __restrict__ ohead) {
  __shared__ bf16 Qs[64 * LDT], Ks[64 * LDT], Vt[64 * LDT], Ps[64 * LDT];
  int tid = threadIdx.x;
  int wave = tid >> 5, lane = tid & 31;
  int hf = lane >> 4, ln = lane & 15;
  int batch = blockIdx.y;
  int q0 = blockIdx.x * 64;
  const bf16* base = qkv + (size_t)batch * SEQ * QKVS;
  const bf16* vTb  = vT  + (size_t)batch * DIMQ * SEQ;
  int row = tid >> 1, h = (tid & 1) * 32;

  { // load Q tile 64x64
    const bf16* src = base + (size_t)(q0 + row) * QKVS + h;
    bf16* dst = &Qs[row * LDT + h];
    #pragma unroll
    for (int j = 0; j < 4; ++j) copy16(src + j * 8, dst + j * 8);
  }
  WAIT_ASYNC(0);
  __syncthreads();

  v16bf aq[2];
  #pragma unroll
  for (int kk = 0; kk < 2; ++kk)
    aq[kk] = load_a_frag(&Qs[(wave * 16) * LDT + kk * 32], LDT, lane);

  float rmax[8], rsum[8];
  #pragma unroll
  for (int i = 0; i < 8; ++i) { rmax[i] = -3.0e38f; rsum[i] = 0.f; }
  v8f o[4] = {};

  for (int kt = 0; kt < SEQ; kt += 64) {
    { // stage K tile (row-major) + V tile (pre-transposed in vT): all contiguous
      const bf16* sk = base + (size_t)(kt + row) * QKVS + 64 + h;
      bf16* dk = &Ks[row * LDT + h];
      const bf16* sv = vTb + (size_t)row * SEQ + kt + h;   // row = e-dim
      bf16* dv = &Vt[row * LDT + h];
      #pragma unroll
      for (int j = 0; j < 4; ++j) {
        copy16(sk + j * 8, dk + j * 8);
        copy16(sv + j * 8, dv + j * 8);
      }
    }
    WAIT_ASYNC(0);
    __syncthreads();

    // S = Q * K^T / 8   (16 rows x 64 keys per wave)
    v8f sc[4];
    #pragma unroll
    for (int nt = 0; nt < 4; ++nt) {
      v8f s = {};
      #pragma unroll
      for (int kk = 0; kk < 2; ++kk) {
        v16bf bk = load_b_frag(&Ks[(nt * 16) * LDT + kk * 32], LDT, lane);
        s = wmma_bf16(aq[kk], bk, s);
      }
      sc[nt] = s * 0.125f;
    }

    // online softmax; row stats live in the per-lane accumulator slots
    #pragma unroll
    for (int i = 0; i < 8; ++i) {
      float mt = fmaxf(fmaxf(sc[0][i], sc[1][i]), fmaxf(sc[2][i], sc[3][i]));
      #pragma unroll
      for (int m = 1; m < 16; m <<= 1) mt = fmaxf(mt, __shfl_xor(mt, m));
      float mn = fmaxf(rmax[i], mt);
      float corr = __expf(rmax[i] - mn);
      rmax[i] = mn;
      float ls = 0.f;
      int prow = wave * 16 + i + hf * 8;
      #pragma unroll
      for (int nt = 0; nt < 4; ++nt) {
        float p = __expf(sc[nt][i] - mn);
        Ps[prow * LDT + nt * 16 + ln] = (bf16)p;
        ls += p;
      }
      #pragma unroll
      for (int m = 1; m < 16; m <<= 1) ls += __shfl_xor(ls, m);
      rsum[i] = rsum[i] * corr + ls;
      #pragma unroll
      for (int et = 0; et < 4; ++et) o[et][i] *= corr;
    }
    asm volatile("s_wait_dscnt 0" ::: "memory");   // P staged in LDS (same wave)

    // O += P * V
    #pragma unroll
    for (int kk = 0; kk < 2; ++kk) {
      v16bf ap = load_a_frag(&Ps[(wave * 16) * LDT + kk * 32], LDT, lane);
      #pragma unroll
      for (int et = 0; et < 4; ++et) {
        v16bf bv = load_b_frag(&Vt[(et * 16) * LDT + kk * 32], LDT, lane);
        o[et] = wmma_bf16(ap, bv, o[et]);
      }
    }
    __syncthreads();
  }

  #pragma unroll
  for (int i = 0; i < 8; ++i) {
    float inv = 1.0f / rsum[i];
    int m = q0 + wave * 16 + i + hf * 8;
    #pragma unroll
    for (int et = 0; et < 4; ++et)
      ohead[((size_t)batch * SEQ + m) * DIMQ + et * 16 + ln] = (bf16)(o[et][i] * inv);
  }
}

// -------------------- orchestration --------------------

extern "C" void kernel_launch(void* const* d_in, const int* in_sizes, int n_in,
                              void* d_out, int out_size, void* d_ws, size_t ws_size,
                              hipStream_t stream) {
  const float* x     = (const float*)d_in[0];
  const float* wq    = (const float*)d_in[1];
  const float* wk    = (const float*)d_in[2];
  const float* wv    = (const float*)d_in[3];
  const float* lin_w = (const float*)d_in[4];
  const float* lin_b = (const float*)d_in[5];
  const float* ln_g  = (const float*)d_in[6];
  const float* ln_b  = (const float*)d_in[7];
  const float* fc1_w = (const float*)d_in[8];
  const float* fc1_b = (const float*)d_in[9];
  const float* fc2_w = (const float*)d_in[10];
  const float* fc2_b = (const float*)d_in[11];

  char* ws = (char*)d_ws;
  size_t off = 0;
  auto alloc = [&](size_t bytes) { size_t o = off; off += (bytes + 255) & ~(size_t)255; return o; };
  bf16*  h     = (bf16*)(ws + alloc((size_t)MTOT * DIM * 2));
  bf16*  qkv   = (bf16*)(ws + alloc((size_t)MTOT * QKVS * 2));
  bf16*  vT    = (bf16*)(ws + alloc((size_t)BATCH * DIMQ * SEQ * 2));
  bf16*  oh    = (bf16*)(ws + alloc((size_t)MTOT * DIMQ * 2));
  float* out1  = (float*)(ws + alloc((size_t)MTOT * DIM * 4));
  bf16*  h2    = (bf16*)(ws + alloc((size_t)MTOT * DIM * 2));
  bf16*  ffn   = (bf16*)(ws + alloc((size_t)MTOT * 4 * DIM * 2));
  bf16*  wqkvt = (bf16*)(ws + alloc((size_t)QKVS * DIM * 2));      // [256][768]
  bf16*  wlint = (bf16*)(ws + alloc((size_t)DIM * DIMQ * 2));      // [768][64]
  bf16*  wfc1t = (bf16*)(ws + alloc((size_t)4 * DIM * DIM * 2));   // [3072][768]
  bf16*  wfc2t = (bf16*)(ws + alloc((size_t)DIM * 4 * DIM * 2));   // [768][3072]

  // weight prep (all transposed to [N][K] bf16)
  pack_qkv_w<<<(QKVS * DIM + 255) / 256, 256, 0, stream>>>(wq, wk, wv, wqkvt);
  fold_lin_w<<<(DIM * DIMQ + 255) / 256, 256, 0, stream>>>(lin_w, wlint);
  cvt_t_bf16<<<(4 * DIM * DIM + 255) / 256, 256, 0, stream>>>(fc1_w, wfc1t, DIM, 4 * DIM);
  cvt_t_bf16<<<(4 * DIM * DIM + 255) / 256, 256, 0, stream>>>(fc2_w, wfc2t, 4 * DIM, DIM);

  // LN1 -> h (bf16)
  layernorm_kernel<<<MTOT, 256, 0, stream>>>(x, ln_g, ln_b, h);

  // fused QKV projection: [8192x768] x [768x256] -> qkv bf16 + vT sideband
  gemm_bf16<3><<<dim3(QKVS / 128, MTOT / 128), 256, 0, stream>>>(
      h, wqkvt, nullptr, nullptr, qkv, vT, MTOT, QKVS, DIM, QKVS);

  // flash attention -> oh bf16 [8192x64]
  attn_kernel<<<dim3(SEQ / 64, BATCH), 128, 0, stream>>>(qkv, vT, oh);

  // folded output proj + bias + residual(x) -> out1 f32
  gemm_bf16<1><<<dim3(DIM / 128, MTOT / 128), 256, 0, stream>>>(
      oh, wlint, lin_b, x, out1, nullptr, MTOT, DIM, DIMQ, DIM);

  // LN (reuses ln1 params) -> h2 bf16
  layernorm_kernel<<<MTOT, 256, 0, stream>>>(out1, ln_g, ln_b, h2);

  // fc1 + GELU -> ffn bf16 [8192x3072]
  gemm_bf16<2><<<dim3(4 * DIM / 128, MTOT / 128), 256, 0, stream>>>(
      h2, wfc1t, fc1_b, nullptr, ffn, nullptr, MTOT, 4 * DIM, DIM, 4 * DIM);

  // fc2 + bias + residual(out1) -> d_out f32
  gemm_bf16<1><<<dim3(DIM / 128, MTOT / 128), 256, 0, stream>>>(
      ffn, wfc2t, fc2_b, out1, (float*)d_out, nullptr, MTOT, DIM, 4 * DIM, DIM);
}
